// QuantumAttentionLayer_7232724927091
// MI455X (gfx1250) — compile-verified, matrix operand order checked by reading, not run on hardware
//
#include <hip/hip_runtime.h>
#include <hip/hip_bf16.h>

// ---------------------------------------------------------------------------
// Types for CDNA5 WMMA (wave32, 16x16x32 bf16 -> f32 accumulate)
// ---------------------------------------------------------------------------
typedef __attribute__((ext_vector_type(16))) __bf16 v16bf;
typedef __attribute__((ext_vector_type(8)))  float  v8f;

#define SEQ   2048
#define HID   1024
#define NHEAD 16
#define HD    64
#define NTOK  4096   // B*S

__device__ __forceinline__ v8f vzero8() {
  v8f z = {0.f, 0.f, 0.f, 0.f, 0.f, 0.f, 0.f, 0.f};
  return z;
}

// WMMA D = A(16x32 bf16) * B(32x16 bf16) + C(16x16 f32)
__device__ __forceinline__ v8f wmma_bf16(v16bf a, v16bf b, v8f c) {
  return __builtin_amdgcn_wmma_f32_16x16x32_bf16(
      false, a, false, b, (short)0, c, false, false);
}

// ISA 7.12.2 16-bit A(16x32)/B(32x16) packed-element K map:
//   K = 16*(e>>3) + 8*(lane>>4) + (e&7);  row/col = lane & 15
__device__ __forceinline__ int kmap(int e, int lane) {
  return ((e >> 3) << 4) + ((lane >> 4) << 3) + (e & 7);
}
// C/D (16x16 f32): element r -> row M = r + 8*(lane>>4), col N = lane & 15.

// ---------------------------------------------------------------------------
// gfx1250 async global->LDS copy (ASYNCcnt-tracked, no VGPR round trip).
// Generic pointers to __shared__ carry the LDS offset in their low 32 bits
// (flat LDS aperture truncates addr[31:0]), which is what VDST expects.
// ---------------------------------------------------------------------------
__device__ __forceinline__ unsigned lds_off(const void* p) {
  return (unsigned)(unsigned long long)p;
}
__device__ __forceinline__ void async_b128(const void* lds_dst, const void* gsrc) {
  asm volatile("global_load_async_to_lds_b128 %0, %1, off"
               :: "v"(lds_off(lds_dst)), "v"(gsrc)
               : "memory");
}
__device__ __forceinline__ void wait_async0() {
  asm volatile("s_wait_asynccnt 0x0" ::: "memory");
}

// ---------------------------------------------------------------------------
// Kernel 0: one-shot f32 -> bf16 convert (x and the four weight matrices)
// ---------------------------------------------------------------------------
__global__ __launch_bounds__(256) void cvt_kernel(
    const float* __restrict__ src, __bf16* __restrict__ dst)
{
  int i = blockIdx.x * 256 + threadIdx.x;
  dst[i] = (__bf16)src[i];
}

// ---------------------------------------------------------------------------
// GEMM tile staging: 128x32 bf16 tile (8 KB) = 512 x b128; 2 chunks/thread.
// ---------------------------------------------------------------------------
__device__ __forceinline__ void stage_pair(
    __bf16 (*Ab)[32], __bf16 (*Bb)[32],
    const __bf16* __restrict__ A, const __bf16* __restrict__ B,
    int row0, int col0, int k0, int tid)
{
#pragma unroll
  for (int i = 0; i < 2; ++i) {
    int chunk = tid + i * 256;
    int r = chunk >> 2, q = (chunk & 3) * 8;
    async_b128(&Ab[r][q], &A[(size_t)(row0 + r) * HID + k0 + q]);
    async_b128(&Bb[r][q], &B[(size_t)(col0 + r) * HID + k0 + q]);
  }
}

// ---------------------------------------------------------------------------
// Kernel 1: fused QKV projection   out = x @ W^T + b   (bf16 in, bf16 out)
// grid: (NTOK/128, HID/128, 3)  block 256 (8 waves: 4 row-groups x 2 col)
// double-buffered async LDS staging overlapped with WMMA
// ---------------------------------------------------------------------------
__global__ __launch_bounds__(256) void qkv_kernel(
    const __bf16* __restrict__ xbf,
    const __bf16* __restrict__ wqbf, const float* __restrict__ bq,
    const __bf16* __restrict__ wkbf, const float* __restrict__ bk,
    const __bf16* __restrict__ wvbf, const float* __restrict__ bv,
    __bf16* __restrict__ qws, __bf16* __restrict__ kws, __bf16* __restrict__ vws)
{
  const int z = blockIdx.z;
  const __bf16* W    = (z == 0) ? wqbf : (z == 1) ? wkbf : wvbf;
  const float*  bias = (z == 0) ? bq   : (z == 1) ? bk   : bv;
  __bf16*       out  = (z == 0) ? qws  : (z == 1) ? kws  : vws;

  __shared__ __bf16 Ab[2][128][32];   // 16 KB
  __shared__ __bf16 Bb[2][128][32];   // 16 KB

  const int tid  = threadIdx.x;
  const int lane = tid & 31;
  const int wave = tid >> 5;
  const int wy   = wave & 3;
  const int wx   = wave >> 2;
  const int ln   = lane & 15;
  const int lh   = lane >> 4;

  const int row0 = blockIdx.x * 128;
  const int col0 = blockIdx.y * 128;

  v8f acc[2][4];
#pragma unroll
  for (int i = 0; i < 2; ++i)
#pragma unroll
    for (int j = 0; j < 4; ++j) acc[i][j] = vzero8();

  stage_pair(Ab[0], Bb[0], xbf, W, row0, col0, 0, tid);
  wait_async0();
  __syncthreads();

  for (int k0 = 0; k0 < HID; k0 += 32) {
    const int bi = (k0 >> 5) & 1;
    if (k0 + 32 < HID)
      stage_pair(Ab[bi ^ 1], Bb[bi ^ 1], xbf, W, row0, col0, k0 + 32, tid);

    v16bf a[2], b[4];
#pragma unroll
    for (int rt = 0; rt < 2; ++rt) {
      int rl = wy * 32 + rt * 16 + ln;
#pragma unroll
      for (int e = 0; e < 16; ++e) a[rt][e] = Ab[bi][rl][kmap(e, lane)];
    }
#pragma unroll
    for (int ct = 0; ct < 4; ++ct) {
      int cl = wx * 64 + ct * 16 + ln;
#pragma unroll
      for (int e = 0; e < 16; ++e) b[ct][e] = Bb[bi][cl][kmap(e, lane)];
    }
#pragma unroll
    for (int rt = 0; rt < 2; ++rt)
#pragma unroll
      for (int ct = 0; ct < 4; ++ct)
        acc[rt][ct] = wmma_bf16(a[rt], b[ct], acc[rt][ct]);

    wait_async0();
    __syncthreads();
  }

  // epilogue: + bias, scatter to [b][h][s][d] bf16
#pragma unroll
  for (int rt = 0; rt < 2; ++rt)
#pragma unroll
    for (int ct = 0; ct < 4; ++ct)
#pragma unroll
      for (int r = 0; r < 8; ++r) {
        int tok = row0 + wy * 32 + rt * 16 + r + lh * 8;
        int n   = col0 + wx * 64 + ct * 16 + ln;
        float v = acc[rt][ct][r] + bias[n];
        int bb = tok >> 11, s = tok & (SEQ - 1);
        int h = n >> 6, d = n & 63;
        out[((size_t)(bb * NHEAD + h) * SEQ + s) * HD + d] = (__bf16)v;
      }
}

// ---------------------------------------------------------------------------
// Kernel 2: fused attention, 3-pass recompute, async double-buffered K/V.
//   scores = QK^T/8 + mask ; p = softmax ; ep = exp(p)
//   p2 = ep/(sum ep + 1e-6) ; cw = p2*sigmoid(p2) ; attn = cw @ V  (bf16 out)
// grid: (SEQ/128, B*NH)  block 256 (8 waves; each wave owns 16 q rows)
// ---------------------------------------------------------------------------
__device__ __forceinline__ void stage_kv32(
    __bf16 (*dst)[HD], const __bf16* __restrict__ src, int k0, int tid)
{
  // 32x64 bf16 tile (4 KB) = 256 x b128, exactly one chunk per thread
  int r = tid >> 3, q = (tid & 7) * 8;
  async_b128(&dst[r][q], &src[(size_t)(k0 + r) * HD + q]);
}

__global__ __launch_bounds__(256) void attn_kernel(
    const __bf16* __restrict__ qws, const __bf16* __restrict__ kws,
    const __bf16* __restrict__ vws, const float* __restrict__ mask,
    __bf16* __restrict__ attnbf)
{
  __shared__ __bf16 Kb[2][32][HD];   // 8 KB
  __shared__ __bf16 Vb[2][32][HD];   // 8 KB
  __shared__ float  CW[8][16][32];   // 16 KB C->A layout bounce

  const int tid  = threadIdx.x;
  const int lane = tid & 31;
  const int wave = tid >> 5;
  const int ln   = lane & 15;
  const int lh   = lane >> 4;

  const int bh = blockIdx.y;
  const int b  = bh >> 4;
  const int h  = bh & (NHEAD - 1);
  const __bf16* Q = qws + (size_t)bh * SEQ * HD;
  const __bf16* K = kws + (size_t)bh * SEQ * HD;
  const __bf16* V = vws + (size_t)bh * SEQ * HD;
  const float*  mrow = mask + (size_t)b * SEQ;

  const int q0 = blockIdx.x * 128 + wave * 16;
  const float scale = 0.125f;

  v16bf qa[2];
#pragma unroll
  for (int c = 0; c < 2; ++c)
#pragma unroll
    for (int e = 0; e < 16; ++e)
      qa[c][e] = Q[(size_t)(q0 + ln) * HD + c * 32 + kmap(e, lane)];

  float m[8], l[8];
#pragma unroll
  for (int r = 0; r < 8; ++r) { m[r] = -1e30f; l[r] = 0.f; }

  // ---------------- pass 1: online row max + softmax denominator ----------
  stage_kv32(Kb[0], K, 0, tid);
  wait_async0();
  __syncthreads();
  for (int k0 = 0; k0 < SEQ; k0 += 32) {
    const int bi = (k0 >> 5) & 1;
    if (k0 + 32 < SEQ) stage_kv32(Kb[bi ^ 1], K, k0 + 32, tid);

    v8f st[2];
#pragma unroll
    for (int t = 0; t < 2; ++t) {
      v16bf bf0, bf1;
#pragma unroll
      for (int e = 0; e < 16; ++e) {
        int kd = kmap(e, lane);
        bf0[e] = Kb[bi][t * 16 + ln][kd];
        bf1[e] = Kb[bi][t * 16 + ln][32 + kd];
      }
      v8f s = vzero8();
      s = wmma_bf16(qa[0], bf0, s);
      s = wmma_bf16(qa[1], bf1, s);
      st[t] = s;
    }
    float m0 = mrow[k0 + ln], m1 = mrow[k0 + 16 + ln];
#pragma unroll
    for (int r = 0; r < 8; ++r) {
      float v0 = st[0][r] * scale + m0;
      float v1 = st[1][r] * scale + m1;
      float mx = fmaxf(v0, v1);
#pragma unroll
      for (int off = 1; off < 16; off <<= 1)
        mx = fmaxf(mx, __shfl_xor(mx, off, 32));
      float mn = fmaxf(m[r], mx);
      float se = __expf(v0 - mn) + __expf(v1 - mn);
#pragma unroll
      for (int off = 1; off < 16; off <<= 1)
        se += __shfl_xor(se, off, 32);
      l[r] = l[r] * __expf(m[r] - mn) + se;
      m[r] = mn;
    }
    wait_async0();
    __syncthreads();
  }
  float linv[8];
#pragma unroll
  for (int r = 0; r < 8; ++r) linv[r] = 1.f / l[r];

  // ---------------- pass 2: sum of exp(p) per row --------------------------
  float sep[8];
#pragma unroll
  for (int r = 0; r < 8; ++r) sep[r] = 0.f;

  stage_kv32(Kb[0], K, 0, tid);
  wait_async0();
  __syncthreads();
  for (int k0 = 0; k0 < SEQ; k0 += 32) {
    const int bi = (k0 >> 5) & 1;
    if (k0 + 32 < SEQ) stage_kv32(Kb[bi ^ 1], K, k0 + 32, tid);

    v8f st[2];
#pragma unroll
    for (int t = 0; t < 2; ++t) {
      v16bf bf0, bf1;
#pragma unroll
      for (int e = 0; e < 16; ++e) {
        int kd = kmap(e, lane);
        bf0[e] = Kb[bi][t * 16 + ln][kd];
        bf1[e] = Kb[bi][t * 16 + ln][32 + kd];
      }
      v8f s = vzero8();
      s = wmma_bf16(qa[0], bf0, s);
      s = wmma_bf16(qa[1], bf1, s);
      st[t] = s;
    }
    float m0 = mrow[k0 + ln], m1 = mrow[k0 + 16 + ln];
#pragma unroll
    for (int r = 0; r < 8; ++r) {
      float p0 = __expf(st[0][r] * scale + m0 - m[r]) * linv[r];
      float p1 = __expf(st[1][r] * scale + m1 - m[r]) * linv[r];
      sep[r] += __expf(p0) + __expf(p1);
    }
    wait_async0();
    __syncthreads();
  }
  float dinv[8];
#pragma unroll
  for (int r = 0; r < 8; ++r) {
    float s = sep[r];
#pragma unroll
    for (int off = 1; off < 16; off <<= 1) s += __shfl_xor(s, off, 32);
    dinv[r] = 1.f / (s + 1e-6f);
  }

  // ---------------- pass 3: gated weights, PV accumulation -----------------
  v8f acc[4];
#pragma unroll
  for (int i = 0; i < 4; ++i) acc[i] = vzero8();

  stage_kv32(Kb[0], K, 0, tid);
  stage_kv32(Vb[0], V, 0, tid);
  wait_async0();
  __syncthreads();
  for (int k0 = 0; k0 < SEQ; k0 += 32) {
    const int bi = (k0 >> 5) & 1;
    if (k0 + 32 < SEQ) {
      stage_kv32(Kb[bi ^ 1], K, k0 + 32, tid);
      stage_kv32(Vb[bi ^ 1], V, k0 + 32, tid);
    }

    float m0 = mrow[k0 + ln], m1 = mrow[k0 + 16 + ln];
#pragma unroll
    for (int t = 0; t < 2; ++t) {
      v16bf bf0, bf1;
#pragma unroll
      for (int e = 0; e < 16; ++e) {
        int kd = kmap(e, lane);
        bf0[e] = Kb[bi][t * 16 + ln][kd];
        bf1[e] = Kb[bi][t * 16 + ln][32 + kd];
      }
      v8f s = vzero8();
      s = wmma_bf16(qa[0], bf0, s);
      s = wmma_bf16(qa[1], bf1, s);
      float madd = t ? m1 : m0;
#pragma unroll
      for (int r = 0; r < 8; ++r) {
        float p  = __expf(s[r] * scale + madd - m[r]) * linv[r];
        float p2 = __expf(p) * dinv[r];
        float cw = p2 / (1.f + __expf(-p2));
        CW[wave][r + 8 * lh][t * 16 + ln] = cw;
      }
    }
    __syncthreads();   // CW visible across the wave halves

    v16bf ca;
#pragma unroll
    for (int e = 0; e < 16; ++e)
      ca[e] = (__bf16)CW[wave][ln][kmap(e, lane)];

#pragma unroll
    for (int ct = 0; ct < 4; ++ct) {
      v16bf vb;
#pragma unroll
      for (int e = 0; e < 16; ++e)
        vb[e] = Vb[bi][kmap(e, lane)][ct * 16 + ln];
      acc[ct] = wmma_bf16(ca, vb, acc[ct]);
    }
    wait_async0();
    __syncthreads();
  }

  // write attn as bf16 [token][h*64+d]  (feeds the async-staged oproj GEMM)
#pragma unroll
  for (int ct = 0; ct < 4; ++ct)
#pragma unroll
    for (int r = 0; r < 8; ++r) {
      int srow = q0 + r + 8 * lh;
      int tok  = b * SEQ + srow;
      int n    = h * HD + ct * 16 + ln;
      attnbf[(size_t)tok * HID + n] = (__bf16)acc[ct][r];
    }
}

// ---------------------------------------------------------------------------
// Kernel 3: output projection + residual:  y = attn @ Wo^T + bo + x  (f32 out)
// ---------------------------------------------------------------------------
__global__ __launch_bounds__(256) void oproj_kernel(
    const __bf16* __restrict__ attn, const __bf16* __restrict__ wobf,
    const float* __restrict__ bo, const float* __restrict__ x,
    float* __restrict__ yws)
{
  __shared__ __bf16 Ab[2][128][32];
  __shared__ __bf16 Bb[2][128][32];

  const int tid  = threadIdx.x;
  const int lane = tid & 31;
  const int wave = tid >> 5;
  const int wy   = wave & 3;
  const int wx   = wave >> 2;
  const int ln   = lane & 15;
  const int lh   = lane >> 4;

  const int row0 = blockIdx.x * 128;
  const int col0 = blockIdx.y * 128;

  v8f acc[2][4];
#pragma unroll
  for (int i = 0; i < 2; ++i)
#pragma unroll
    for (int j = 0; j < 4; ++j) acc[i][j] = vzero8();

  stage_pair(Ab[0], Bb[0], attn, wobf, row0, col0, 0, tid);
  wait_async0();
  __syncthreads();

  for (int k0 = 0; k0 < HID; k0 += 32) {
    const int bi = (k0 >> 5) & 1;
    if (k0 + 32 < HID)
      stage_pair(Ab[bi ^ 1], Bb[bi ^ 1], attn, wobf, row0, col0, k0 + 32, tid);

    v16bf a[2], bfr[4];
#pragma unroll
    for (int rt = 0; rt < 2; ++rt) {
      int rl = wy * 32 + rt * 16 + ln;
#pragma unroll
      for (int e = 0; e < 16; ++e) a[rt][e] = Ab[bi][rl][kmap(e, lane)];
    }
#pragma unroll
    for (int ct = 0; ct < 4; ++ct) {
      int cl = wx * 64 + ct * 16 + ln;
#pragma unroll
      for (int e = 0; e < 16; ++e) bfr[ct][e] = Bb[bi][cl][kmap(e, lane)];
    }
#pragma unroll
    for (int rt = 0; rt < 2; ++rt)
#pragma unroll
      for (int ct = 0; ct < 4; ++ct)
        acc[rt][ct] = wmma_bf16(a[rt], bfr[ct], acc[rt][ct]);

    wait_async0();
    __syncthreads();
  }

#pragma unroll
  for (int rt = 0; rt < 2; ++rt)
#pragma unroll
    for (int ct = 0; ct < 4; ++ct)
#pragma unroll
      for (int r = 0; r < 8; ++r) {
        int tok = row0 + wy * 32 + rt * 16 + r + lh * 8;
        int n   = col0 + wx * 64 + ct * 16 + ln;
        size_t off = (size_t)tok * HID + n;
        yws[off] = acc[rt][ct][r] + bo[n] + x[off];
      }
}

// ---------------------------------------------------------------------------
// Kernel 4: LayerNorm with Quake-III inverse sqrt (exact bit trick)
// ---------------------------------------------------------------------------
__device__ __forceinline__ float fast_inv_sqrt(float x) {
  float x2 = x * 0.5f;
  int i = __float_as_int(x);
  i = 1597463007 - (i >> 1);
  float y0 = __int_as_float(i);
  return y0 * (1.5f - x2 * y0 * y0);
}

__global__ __launch_bounds__(256) void ln_kernel(
    const float* __restrict__ yws, const float* __restrict__ gw,
    const float* __restrict__ gb, float* __restrict__ out)
{
  const int tok = blockIdx.x;
  const float* y = yws + (size_t)tok * HID;
  float vals[4], s = 0.f, s2 = 0.f;
#pragma unroll
  for (int i = 0; i < 4; ++i) {
    float v = y[threadIdx.x + i * 256];
    vals[i] = v; s += v; s2 += v * v;
  }
#pragma unroll
  for (int off = 1; off < 32; off <<= 1) {
    s  += __shfl_xor(s,  off, 32);
    s2 += __shfl_xor(s2, off, 32);
  }
  __shared__ float red[16];
  __shared__ float mean_s, inv_s;
  int wave = threadIdx.x >> 5, lane = threadIdx.x & 31;
  if (lane == 0) { red[wave] = s; red[8 + wave] = s2; }
  __syncthreads();
  if (threadIdx.x == 0) {
    float ts = 0.f, ts2 = 0.f;
#pragma unroll
    for (int w = 0; w < 8; ++w) { ts += red[w]; ts2 += red[8 + w]; }
    float mean = ts * (1.f / HID);
    float var  = ts2 * (1.f / HID) - mean * mean;
    mean_s = mean;
    inv_s  = fast_inv_sqrt(var + 1e-5f);
  }
  __syncthreads();
  float mean = mean_s, inv = inv_s;
#pragma unroll
  for (int i = 0; i < 4; ++i) {
    int c = threadIdx.x + i * 256;
    out[(size_t)tok * HID + c] = gw[c] * ((vals[i] - mean) * inv) + gb[c];
  }
}

// ---------------------------------------------------------------------------
// Host launcher
// ---------------------------------------------------------------------------
extern "C" void kernel_launch(void* const* d_in, const int* in_sizes, int n_in,
                              void* d_out, int out_size, void* d_ws, size_t ws_size,
                              hipStream_t stream) {
  const float* hs   = (const float*)d_in[0];
  const float* mask = (const float*)d_in[1];
  const float* Wq   = (const float*)d_in[2];
  const float* bq   = (const float*)d_in[3];
  const float* Wk   = (const float*)d_in[4];
  const float* bk   = (const float*)d_in[5];
  const float* Wv   = (const float*)d_in[6];
  const float* bv   = (const float*)d_in[7];
  const float* Wo   = (const float*)d_in[8];
  const float* bo   = (const float*)d_in[9];
  const float* nw   = (const float*)d_in[10];
  const float* nb   = (const float*)d_in[11];

  // ws: xbf(4M) | wq/wk/wv/wo bf(1M each) | q/k/v bf(4M each) | attn bf(4M) | y f32
  __bf16* xbf  = (__bf16*)d_ws;
  __bf16* wqbf = xbf  + (size_t)NTOK * HID;
  __bf16* wkbf = wqbf + (size_t)HID * HID;
  __bf16* wvbf = wkbf + (size_t)HID * HID;
  __bf16* wobf = wvbf + (size_t)HID * HID;
  __bf16* qws  = wobf + (size_t)HID * HID;
  __bf16* kws  = qws  + (size_t)NTOK * HID;
  __bf16* vws  = kws  + (size_t)NTOK * HID;
  __bf16* attnbf = vws + (size_t)NTOK * HID;
  float*  yws  = (float*)(attnbf + (size_t)NTOK * HID);

  dim3 blk(256);
  cvt_kernel<<<dim3(NTOK * HID / 256), blk, 0, stream>>>(hs, xbf);
  cvt_kernel<<<dim3(HID * HID / 256), blk, 0, stream>>>(Wq, wqbf);
  cvt_kernel<<<dim3(HID * HID / 256), blk, 0, stream>>>(Wk, wkbf);
  cvt_kernel<<<dim3(HID * HID / 256), blk, 0, stream>>>(Wv, wvbf);
  cvt_kernel<<<dim3(HID * HID / 256), blk, 0, stream>>>(Wo, wobf);

  qkv_kernel<<<dim3(NTOK / 128, HID / 128, 3), blk, 0, stream>>>(
      xbf, wqbf, bq, wkbf, bk, wvbf, bv, qws, kws, vws);
  attn_kernel<<<dim3(SEQ / 128, 2 * NHEAD), blk, 0, stream>>>(
      qws, kws, vws, mask, attnbf);
  oproj_kernel<<<dim3(NTOK / 128, HID / 128), blk, 0, stream>>>(
      attnbf, wobf, bo, hs, yws);
  ln_kernel<<<dim3(NTOK), blk, 0, stream>>>(yws, nw, nb, (float*)d_out);
}
